// tsie_77910706749507
// MI455X (gfx1250) — matrix operand order repaired and checked by reading, not compiled, not run on hardware
//
#include <hip/hip_runtime.h>

// ---------------------------------------------------------------------------
// Model constants (from reference)
// ---------------------------------------------------------------------------
#define T16   16
#define NN    10000
#define NB    1024
#define FIN   128
#define HH    256
#define DD    512
#define H3    768
#define NEDGE 80000
#define WND   8
#define NHEAD 8
#define DH    64

typedef __attribute__((ext_vector_type(16))) _Float16 v16h;
typedef __attribute__((ext_vector_type(8)))  float    v8f;

struct alignas(16) U128 { unsigned int x, y, z, w; };

__device__ __forceinline__ float wave_red(float v) {
#pragma unroll
  for (int m = 16; m > 0; m >>= 1) v += __shfl_xor(v, m, 32);
  return v;
}
__device__ __forceinline__ float sigmoidf(float x) { return 1.0f / (1.0f + expf(-x)); }

// ---------------------------------------------------------------------------
// Utility kernels
// ---------------------------------------------------------------------------
__global__ void k_zero(float4* __restrict__ p, long long n4) {
  long long id = (long long)blockIdx.x * blockDim.x + threadIdx.x;
  if (id < n4) p[id] = make_float4(0.f, 0.f, 0.f, 0.f);
}

__global__ void k_cvt_f16(const float* __restrict__ s, _Float16* __restrict__ d, int n) {
  int id = blockIdx.x * blockDim.x + threadIdx.x;
  if (id < n) d[id] = (_Float16)s[id];
}

// Build sage weight as (N=256, K=256) f16: k<128 -> W_self[k][n], else W_neigh[k-128][n]
__global__ void k_sage_w16(const float* __restrict__ wself, const float* __restrict__ wneigh,
                           _Float16* __restrict__ d) {
  int id = blockIdx.x * blockDim.x + threadIdx.x;
  if (id >= 256 * 256) return;
  int n = id >> 8, k = id & 255;
  float v = (k < 128) ? wself[k * HH + n] : wneigh[(k - 128) * HH + n];
  d[n * 256 + k] = (_Float16)v;
}

// ---------------------------------------------------------------------------
// Stage A: graph aggregation (scatter-add), feature assembly
// ---------------------------------------------------------------------------
__global__ void k_scatter(const float* __restrict__ xi, const float* __restrict__ xj,
                          const int* __restrict__ ei, const int* __restrict__ ej,
                          float* __restrict__ agg, float* __restrict__ deg) {
  long long id = (long long)blockIdx.x * blockDim.x + threadIdx.x;
  int chunk = (int)(id & 31);
  long long r = id >> 5;
  if (r >= (long long)2 * T16 * NEDGE) return;
  int e = (int)(r % NEDGE);
  int t = (int)((r / NEDGE) % T16);
  int g = (int)(r / ((long long)NEDGE * T16));
  const int* E = (g ? ej : ei) + ((long long)t * NEDGE + e) * 2;
  int src = E[0], dst = E[1];
  const float* xp = (g ? xj : xi) + ((size_t)t * NN + src) * FIN + chunk * 4;
  float4 xv = *(const float4*)xp;
  float* ap = agg + ((size_t)(g * T16 + t) * NN + dst) * FIN + chunk * 4;
  atomicAdd(ap + 0, xv.x); atomicAdd(ap + 1, xv.y);
  atomicAdd(ap + 2, xv.z); atomicAdd(ap + 3, xv.w);
  if (chunk == 0) atomicAdd(deg + (size_t)(g * T16 + t) * NN + dst, 1.0f);
}

// Acat[(g,t,b)][0:128]=x[t,node], [128:256]=agg/deg   (rows = 32768, cols = 256)
__global__ void k_acat(const float* __restrict__ xi, const float* __restrict__ xj,
                       const int* __restrict__ ii, const int* __restrict__ ij,
                       const float* __restrict__ agg, const float* __restrict__ deg,
                       float* __restrict__ acat) {
  int id = blockIdx.x * blockDim.x + threadIdx.x;
  if (id >= 32768 * 64) return;
  int row = id >> 6, c4 = id & 63;
  int g = row >> 14, t = (row >> 10) & 15, b = row & 1023;
  int node = (g ? ij : ii)[t * NB + b];
  float4 o;
  if (c4 < 32) {
    const float* xp = (g ? xj : xi) + ((size_t)t * NN + node) * FIN + c4 * 4;
    o = *(const float4*)xp;
  } else {
    size_t base = (size_t)(g * T16 + t) * NN + node;
    float dv = fmaxf(deg[base], 1.0f);
    float4 a = *(const float4*)(agg + base * FIN + (c4 - 32) * 4);
    o = make_float4(a.x / dv, a.y / dv, a.z / dv, a.w / dv);
  }
  *(float4*)(acat + (size_t)row * 256 + c4 * 4) = o;
}

__global__ void k_rownorm(float* __restrict__ h, int rows) {
  int wid = threadIdx.x >> 5, lane = threadIdx.x & 31;
  int row = blockIdx.x * 8 + wid;
  if (row >= rows) return;
  float* p = h + (size_t)row * HH;
  float s = 0.f;
#pragma unroll
  for (int k = 0; k < 8; k++) { float v = p[lane + 32 * k]; s += v * v; }
  s = wave_red(s);
  float inv = 1.0f / fmaxf(sqrtf(s), 1e-12f);
#pragma unroll
  for (int k = 0; k < 8; k++) p[lane + 32 * k] *= inv;
}

// ---------------------------------------------------------------------------
// WMMA f16 GEMM:  C[M,N] = act( A[M,K](f32) @ W[N,K](f16)^T + bias[N] )
// M%128==0, N%128==0, K%32==0.  Block 256 threads = 8 waves, tile 128x128.
// A: f32 -> f16 through double-buffered LDS (register staged, 1 barrier/step).
// B: fragments loaded DIRECTLY from global (weights are K-contiguous in the
//    same k = 16h + s order as A; shared across blocks -> L2 resident). No B LDS.
// ---------------------------------------------------------------------------
#define BM  128
#define BN  128
#define BK  32
#define LDT 40  // LDS row stride in halfs (pad to dodge bank conflicts)

__launch_bounds__(256)
__global__ void k_gemm(const float* __restrict__ A, const _Float16* __restrict__ W,
                       const float* __restrict__ bias, float* __restrict__ C,
                       int M, int N, int K, int act) {
  __shared__ _Float16 As[2][BM * LDT];
  const int tid  = threadIdx.x;
  const int wid  = tid >> 5, lane = tid & 31;
  const int wm   = wid & 1, wn = wid >> 1;        // 2 x 4 wave grid
  const int mBase = blockIdx.y * BM;
  const int nBase = blockIdx.x * BN;
  const int lm = lane & 15, lh = lane >> 4;

  v8f acc[4][2];
#pragma unroll
  for (int mi = 0; mi < 4; mi++)
#pragma unroll
    for (int ni = 0; ni < 2; ni++)
#pragma unroll
      for (int e = 0; e < 8; e++) acc[mi][ni][e] = 0.0f;

  // per-thread advancing pointers (A staging quads, B fragment rows)
  const int r0 = tid >> 3, q0 = tid & 7;          // A quad decomposition
  const float* aP[4];
#pragma unroll
  for (int it = 0; it < 4; it++)
    aP[it] = A + (size_t)(mBase + r0 + it * 32) * K + q0 * 4;
  const _Float16* wP[2];
#pragma unroll
  for (int ni = 0; ni < 2; ni++)
    wP[ni] = W + (size_t)(nBase + wn * 32 + ni * 16 + lm) * K + lh * 16;

  float4 ar[4];                                   // A staging registers
  auto load_A = [&]() {
#pragma unroll
    for (int it = 0; it < 4; it++) { ar[it] = *(const float4*)aP[it]; aP[it] += BK; }
  };
  auto store_A = [&](int buf) {
#pragma unroll
    for (int it = 0; it < 4; it++) {
      union { _Float16 h[4]; unsigned long long u; } p;
      p.h[0] = (_Float16)ar[it].x; p.h[1] = (_Float16)ar[it].y;
      p.h[2] = (_Float16)ar[it].z; p.h[3] = (_Float16)ar[it].w;
      *(unsigned long long*)(&As[buf][(r0 + it * 32) * LDT + q0 * 4]) = p.u;
    }
  };

  const int nk = K / BK;
  load_A();
  store_A(0);

  for (int ki = 0; ki < nk; ki++) {
    const int cur = ki & 1;
    if (ki + 1 < nk) load_A();                    // next A tile: issued first

    // ---- B fragments for this step, straight from global (two b128 per frag)
    v16h bf[2];
#pragma unroll
    for (int ni = 0; ni < 2; ni++) {
      union { U128 u[2]; v16h v; } fb;
      fb.u[0] = *(const U128*)(wP[ni]);
      fb.u[1] = *(const U128*)(wP[ni] + 8);
      wP[ni] += BK;
      bf[ni] = fb.v;
    }

    __syncthreads();                              // A tile `cur` visible; prior reads done

    // ---- A fragments from LDS: halfs[s] = A[m][16h + s]
    v16h af[4];
#pragma unroll
    for (int mi = 0; mi < 4; mi++) {
      const _Float16* base = &As[cur][(wm * 64 + mi * 16 + lm) * LDT + lh * 16];
      union { U128 u[2]; v16h v; } fa;
      fa.u[0] = *(const U128*)(base);
      fa.u[1] = *(const U128*)(base + 8);
      af[mi] = fa.v;
    }

    // drain A staging into the other buffer (waits only the 4 A loads)
    if (ki + 1 < nk) store_A(cur ^ 1);

#pragma unroll
    for (int mi = 0; mi < 4; mi++)
#pragma unroll
      for (int ni = 0; ni < 2; ni++)
        acc[mi][ni] = __builtin_amdgcn_wmma_f32_16x16x32_f16(
            false, af[mi], false, bf[ni], (short)0, acc[mi][ni], false, false);
  }

  // ---- epilogue: C/D layout row = v + 8*(lane/16), col = lane%16
#pragma unroll
  for (int ni = 0; ni < 2; ni++) {
    int col = nBase + wn * 32 + ni * 16 + lm;
    float bv = bias[col];
#pragma unroll
    for (int mi = 0; mi < 4; mi++) {
#pragma unroll
      for (int e = 0; e < 8; e++) {
        int row = mBase + wm * 64 + mi * 16 + lh * 8 + e;
        float vv = acc[mi][ni][e] + bv;
        if (act == 1) vv = fmaxf(vv, 0.0f);
        C[(size_t)row * N + col] = vv;
      }
    }
  }
}

// ---------------------------------------------------------------------------
// GRU step kernels
// ---------------------------------------------------------------------------
// X rows ordered (s,t,b); set 0: {v[t]-u[tp], u[t]-v[tp]}, set 1: {u[tp], v[tp]}
__global__ void k_gru_x(const float* __restrict__ u, const float* __restrict__ v,
                        float* __restrict__ X, int set, int w) {
  int id = blockIdx.x * blockDim.x + threadIdx.x;
  if (id >= 32768 * 64) return;
  int row = id >> 6, c4 = id & 63;
  int s = row >> 14, t = (row >> 10) & 15, b = row & 1023;
  int start = t - 7; if (start < 0) start = 0;
  int tp = start + w;
  size_t cur = ((size_t)(t * NB + b)) * HH + c4 * 4;
  size_t win = ((size_t)(tp * NB + b)) * HH + c4 * 4;
  float4 o;
  if (set == 0) {
    float4 a = s ? *(const float4*)(u + cur) : *(const float4*)(v + cur);
    float4 c = s ? *(const float4*)(v + win) : *(const float4*)(u + win);
    o = make_float4(a.x - c.x, a.y - c.y, a.z - c.z, a.w - c.w);
  } else {
    o = s ? *(const float4*)(v + win) : *(const float4*)(u + win);
  }
  *(float4*)(X + (size_t)row * HH + c4 * 4) = o;
}

__global__ void k_gru_upd(const float* __restrict__ gi, const float* __restrict__ gh,
                          float* __restrict__ h, int w) {
  int id = blockIdx.x * blockDim.x + threadIdx.x;
  if (id >= 32768 * 256) return;
  int row = id >> 8, c = id & 255;
  int t = (row >> 10) & 15;
  int start = t - 7; if (start < 0) start = 0;
  if (start + w > t) return;                 // invalid step: keep h
  size_t gb = (size_t)row * H3 + c;
  float ir = gi[gb], iz = gi[gb + 256], in = gi[gb + 512];
  float hr = gh[gb], hz = gh[gb + 256], hn = gh[gb + 512];
  float r = sigmoidf(ir + hr);
  float z = sigmoidf(iz + hz);
  float n = tanhf(in + r * hn);
  size_t hb = (size_t)row * HH + c;
  float hv = h[hb];
  h[hb] = (1.0f - z) * n + z * hv;
}

// outs = relu([h_q0 | h_q1]);  his = relu([h_q2 | h_q3])
__global__ void k_concat(const float* __restrict__ hbuf, float* __restrict__ outs,
                         float* __restrict__ his) {
  int id = blockIdx.x * blockDim.x + threadIdx.x;
  if (id >= 16384 * 512) return;
  int row = id >> 9, c = id & 511;
  int q = c >> 8, col = c & 255;
  outs[id] = fmaxf(hbuf[((size_t)(q * 16384 + row)) * HH + col], 0.0f);
  his[id]  = fmaxf(hbuf[((size_t)((q + 2) * 16384 + row)) * HH + col], 0.0f);
}

// ---------------------------------------------------------------------------
// Transformer pieces
// ---------------------------------------------------------------------------
// Banded causal attention: one wave per (b,head); lanes 0..15 = query rows.
__global__ void k_attn(const float* __restrict__ qkv, float* __restrict__ o) {
  int wid = threadIdx.x >> 5, lane = threadIdx.x & 31;
  int pair = blockIdx.x * 8 + wid;
  if (pair >= NB * NHEAD || lane >= 16) return;
  int b = pair >> 3, hh = pair & 7;
  int i = lane;
  const float* qrow = qkv + ((size_t)(i * NB + b)) * 1536 + hh * DH;
  float q[DH];
#pragma unroll
  for (int d = 0; d < DH; d++) q[d] = qrow[d] * 0.125f;   // dh^-0.5
  float sc[T16];
#pragma unroll
  for (int j = 0; j < T16; j++) {
    if (j <= i && i - j < WND) {
      const float* krow = qkv + ((size_t)(j * NB + b)) * 1536 + DD + hh * DH;
      float s = 0.f;
#pragma unroll
      for (int d = 0; d < DH; d++) s += q[d] * krow[d];
      sc[j] = s;
    } else sc[j] = -1e9f;
  }
  float m = sc[0];
#pragma unroll
  for (int j = 1; j < T16; j++) m = fmaxf(m, sc[j]);
  float den = 0.f;
#pragma unroll
  for (int j = 0; j < T16; j++) { sc[j] = expf(sc[j] - m); den += sc[j]; }
  float inv = 1.0f / den;
  float acc[DH];
#pragma unroll
  for (int d = 0; d < DH; d++) acc[d] = 0.f;
#pragma unroll
  for (int j = 0; j < T16; j++) {
    float wgt = sc[j] * inv;
    const float* vrow = qkv + ((size_t)(j * NB + b)) * 1536 + 2 * DD + hh * DH;
#pragma unroll
    for (int d = 0; d < DH; d++) acc[d] += wgt * vrow[d];
  }
  float* op = o + ((size_t)(i * NB + b)) * DD + hh * DH;
#pragma unroll
  for (int d = 0; d < DH; d++) op[d] = acc[d];
}

// x = LN(x + h) * gamma + beta  (rows of 512, one wave per row)
__global__ void k_addln(float* __restrict__ x, const float* __restrict__ h,
                        const float* __restrict__ gma, const float* __restrict__ bta) {
  int wid = threadIdx.x >> 5, lane = threadIdx.x & 31;
  int row = blockIdx.x * 8 + wid;
  if (row >= 16384) return;
  float* xp = x + (size_t)row * DD;
  const float* hp = h + (size_t)row * DD;
  float s[16];
  float sum = 0.f;
#pragma unroll
  for (int k = 0; k < 16; k++) { s[k] = xp[lane + 32 * k] + hp[lane + 32 * k]; sum += s[k]; }
  sum = wave_red(sum);
  float mean = sum * (1.0f / 512.0f);
  float var = 0.f;
#pragma unroll
  for (int k = 0; k < 16; k++) { float d = s[k] - mean; var += d * d; }
  var = wave_red(var) * (1.0f / 512.0f);
  float rs = rsqrtf(var + 1e-5f);
#pragma unroll
  for (int k = 0; k < 16; k++) {
    int c = lane + 32 * k;
    xp[c] = (s[k] - mean) * rs * gma[c] + bta[c];
  }
}

// out[row] = dot( sig(gate)*relu(outs) + (1-sig)*relu(his), out_W )
__global__ void k_fuse_out(const float* __restrict__ outs, const float* __restrict__ his,
                           const float* __restrict__ gate, const float* __restrict__ outW,
                           float* __restrict__ out) {
  int wid = threadIdx.x >> 5, lane = threadIdx.x & 31;
  int row = blockIdx.x * 8 + wid;
  if (row >= 16384) return;
  size_t rb = (size_t)row * DD;
  float acc = 0.f;
#pragma unroll
  for (int k = 0; k < 16; k++) {
    int c = lane + 32 * k;
    float g = sigmoidf(gate[rb + c]);
    float f = g * fmaxf(outs[rb + c], 0.f) + (1.0f - g) * fmaxf(his[rb + c], 0.f);
    acc += f * outW[c];
  }
  acc = wave_red(acc);
  if (lane == 0) out[row] = acc;
}

// ---------------------------------------------------------------------------
// Orchestration
// ---------------------------------------------------------------------------
extern "C" void kernel_launch(void* const* d_in, const int* in_sizes, int n_in,
                              void* d_out, int out_size, void* d_ws, size_t ws_size,
                              hipStream_t stream) {
  (void)in_sizes; (void)n_in; (void)out_size; (void)ws_size;

  const float* x_i = (const float*)d_in[0];
  const float* x_j = (const float*)d_in[1];
  const int* edges_i = (const int*)d_in[2];
  const int* edges_j = (const int*)d_in[3];
  const int* idx_i = (const int*)d_in[4];
  const int* idx_j = (const int*)d_in[5];
  const float* sage_Wself  = (const float*)d_in[6];
  const float* sage_Wneigh = (const float*)d_in[7];
  const float* sage_b      = (const float*)d_in[8];
  const float* gru_Wih[2] = {(const float*)d_in[9],  (const float*)d_in[13]};
  const float* gru_Whh[2] = {(const float*)d_in[10], (const float*)d_in[14]};
  const float* gru_bih[2] = {(const float*)d_in[11], (const float*)d_in[15]};
  const float* gru_bhh[2] = {(const float*)d_in[12], (const float*)d_in[16]};
  const int tbase[4] = {17, 29, 41, 53};  // trans1 L0,L1 ; trans2 L0,L1
  const float* fus_W = (const float*)d_in[65];
  const float* fus_b = (const float*)d_in[66];
  const float* outW  = (const float*)d_in[67];
  float* out = (float*)d_out;

  // ---- workspace carve-up (deterministic, ~497 MB)
  char* ws = (char*)d_ws;
  size_t off = 0;
  auto take = [&](size_t bytes) -> char* {
    char* p = ws + off;
    off += (bytes + 255) & ~(size_t)255;
    return p;
  };
  _Float16* wSage = (_Float16*)take((size_t)256 * 256 * 2);
  _Float16 *wIh16[2], *wHh16[2];
  for (int s = 0; s < 2; s++) {
    wIh16[s] = (_Float16*)take((size_t)H3 * HH * 2);
    wHh16[s] = (_Float16*)take((size_t)H3 * HH * 2);
  }
  _Float16 *wIn16[4], *wOut16[4], *wL116[4], *wL216[4];
  for (int l = 0; l < 4; l++) {
    wIn16[l]  = (_Float16*)take((size_t)1536 * 512 * 2);
    wOut16[l] = (_Float16*)take((size_t)512 * 512 * 2);
    wL116[l]  = (_Float16*)take((size_t)2048 * 512 * 2);
    wL216[l]  = (_Float16*)take((size_t)512 * 2048 * 2);
  }
  _Float16* wFus = (_Float16*)take((size_t)512 * 512 * 2);

  float* embOut = (float*)take((size_t)32768 * 256 * 4);   // u | v
  float* hbuf   = (float*)take((size_t)65536 * 256 * 4);   // 4 GRU hidden seqs
  float* outs   = (float*)take((size_t)16384 * 512 * 4);
  float* his    = (float*)take((size_t)16384 * 512 * 4);
  char*  scr    = take((size_t)301989888);                 // shared scratch

  // stage-A views
  float* agg  = (float*)scr;                       // 163,840,000 B
  float* deg  = (float*)(scr + 163840000);         //   1,280,000 B
  float* acat = (float*)(scr + 165120000);         //  33,554,432 B
  // GRU views
  float* Xb  = (float*)scr;
  float* gib = (float*)(scr + 33554432);
  float* ghb = (float*)(scr + 134217728);
  // transformer views
  float* qkvb = (float*)scr;
  float* attb = (float*)(scr + 100663296);
  float* resb = (float*)(scr + 134217728);
  float* ffb  = (float*)(scr + 167772160);

  float* u = embOut;
  float* v = embOut + (size_t)16384 * 256;

  auto cvt = [&](const float* s, _Float16* d, int n) {
    k_cvt_f16<<<(n + 255) / 256, 256, 0, stream>>>(s, d, n);
  };

  // ---- 0. weight conversions (f32 -> f16, (N,K) row-major)
  k_sage_w16<<<256, 256, 0, stream>>>(sage_Wself, sage_Wneigh, wSage);
  for (int s = 0; s < 2; s++) {
    cvt(gru_Wih[s], wIh16[s], H3 * HH);
    cvt(gru_Whh[s], wHh16[s], H3 * HH);
  }
  for (int l = 0; l < 4; l++) {
    cvt((const float*)d_in[tbase[l] + 0], wIn16[l],  1536 * 512);
    cvt((const float*)d_in[tbase[l] + 2], wOut16[l], 512 * 512);
    cvt((const float*)d_in[tbase[l] + 4], wL116[l],  2048 * 512);
    cvt((const float*)d_in[tbase[l] + 6], wL216[l],  512 * 2048);
  }
  cvt(fus_W, wFus, 512 * 512);

  // ---- 1. SAGE embed
  {
    long long n4 = 165120000 / 16;  // agg + deg (contiguous)
    k_zero<<<(unsigned)((n4 + 255) / 256), 256, 0, stream>>>((float4*)agg, n4);
  }
  k_scatter<<<320000, 256, 0, stream>>>(x_i, x_j, edges_i, edges_j, agg, deg);
  k_acat<<<8192, 256, 0, stream>>>(x_i, x_j, idx_i, idx_j, agg, deg, acat);
  k_gemm<<<dim3(2, 256), 256, 0, stream>>>(acat, wSage, sage_b, embOut, 32768, 256, 256, 1);
  k_rownorm<<<4096, 256, 0, stream>>>(embOut, 32768);

  // ---- 2. GRU scans (h init = 0)
  {
    long long n4 = ((size_t)65536 * 256 * 4) / 16;
    k_zero<<<(unsigned)((n4 + 255) / 256), 256, 0, stream>>>((float4*)hbuf, n4);
  }
  for (int set = 0; set < 2; set++) {
    float* hA = hbuf + (size_t)set * 32768 * 256;
    for (int w = 0; w < 8; w++) {
      k_gru_x<<<8192, 256, 0, stream>>>(u, v, Xb, set, w);
      k_gemm<<<dim3(6, 256), 256, 0, stream>>>(Xb, wIh16[set], gru_bih[set], gib,
                                               32768, 768, 256, 0);
      k_gemm<<<dim3(6, 256), 256, 0, stream>>>(hA, wHh16[set], gru_bhh[set], ghb,
                                               32768, 768, 256, 0);
      k_gru_upd<<<32768, 256, 0, stream>>>(gib, ghb, hA, w);
    }
  }
  k_concat<<<32768, 256, 0, stream>>>(hbuf, outs, his);

  // ---- 3. transformer encoders (trans1 on outs, trans2 on his)
  float* streams[2] = {outs, his};
  for (int st = 0; st < 2; st++) {
    float* x = streams[st];
    for (int li = 0; li < 2; li++) {
      int l = st * 2 + li;
      const float* in_b  = (const float*)d_in[tbase[l] + 1];
      const float* out_b = (const float*)d_in[tbase[l] + 3];
      const float* l1_b  = (const float*)d_in[tbase[l] + 5];
      const float* l2_b  = (const float*)d_in[tbase[l] + 7];
      const float* ln1g  = (const float*)d_in[tbase[l] + 8];
      const float* ln1b  = (const float*)d_in[tbase[l] + 9];
      const float* ln2g  = (const float*)d_in[tbase[l] + 10];
      const float* ln2b  = (const float*)d_in[tbase[l] + 11];
      k_gemm<<<dim3(12, 128), 256, 0, stream>>>(x, wIn16[l], in_b, qkvb, 16384, 1536, 512, 0);
      k_attn<<<1024, 256, 0, stream>>>(qkvb, attb);
      k_gemm<<<dim3(4, 128), 256, 0, stream>>>(attb, wOut16[l], out_b, resb, 16384, 512, 512, 0);
      k_addln<<<2048, 256, 0, stream>>>(x, resb, ln1g, ln1b);
      k_gemm<<<dim3(16, 128), 256, 0, stream>>>(x, wL116[l], l1_b, ffb, 16384, 2048, 512, 1);
      k_gemm<<<dim3(4, 128), 256, 0, stream>>>(ffb, wL216[l], l2_b, resb, 16384, 512, 2048, 0);
      k_addln<<<2048, 256, 0, stream>>>(x, resb, ln2g, ln2b);
    }
  }

  // ---- 4. fusion gate + output projection
  k_gemm<<<dim3(4, 128), 256, 0, stream>>>(outs, wFus, fus_b, resb, 16384, 512, 512, 0);
  k_fuse_out<<<2048, 256, 0, stream>>>(outs, his, resb, outW, out);
}
